// LGNN_71390946394351
// MI455X (gfx1250) — compile-verified, matrix operand order checked by reading, not run on hardware
//
#include <hip/hip_runtime.h>

// ---------------------------------------------------------------------------
// LGNN forward for MI455X (gfx1250, wave32).
// GEMMs: bf16 WMMA (v_wmma_f32_16x16x32_bf16), f32 accumulate.
//  - Weights pre-packed once: Wt[n][k] = (bf16)W[k][n]  -> B fragment = 2x b128
//  - Activations staged in bf16                          -> A fragment = 2x b128
//  - 16x64 tile per wave: 1 A fragment reused by 4 WMMAs per K-step
//  - WGP-scope prefetch on the A stream
//  - Epilogue: bias + optional residual addend + optional ReLU, f32 or bf16 out
// Attention/segment-softmax: one wave32 per (edge,head), shfl_xor reduction,
// global f32 atomics. Residual+LayerNorm fused, one wave per row.
// ---------------------------------------------------------------------------

#define N_NODES 10000
#define N_EDGES 160000
#define N_LGE   1000000
#define NDIM    512
#define NHEAD   8
#define HDIM    64      // DK == EDIM == 64
#define FF_N    2048
#define FF_E    256

typedef __attribute__((ext_vector_type(16))) __bf16 v16bf;
typedef __attribute__((ext_vector_type(8)))  __bf16 v8bf;
typedef __attribute__((ext_vector_type(4)))  __bf16 v4bf;
typedef __attribute__((ext_vector_type(8)))  float  v8f;

union BF16Frag { v16bf f; v8bf h[2]; };

// ---------------------------------------------------------------------------
// WMMA GEMM: out[M x Nout] = act( A'[M x K](bf16) @ Wt[Nout x K](bf16)^T
//                                 + bias + addC )
//   A' = gidx ? A[gidx[m]] : A[m]
// One wave computes a 16(M) x 64(N) tile: 4 accumulators, 4 WMMA per K-step.
// Fragment chunks per ISA 7.12.2: lane (r16 = lane&15, hh = lane>>4) holds
// K in [kk+8*hh, +8) and [kk+16+8*hh, +8)  -> two 16B contiguous loads.
// Tile guard is wave-uniform -> EXEC all-ones at every WMMA.
// ---------------------------------------------------------------------------
__global__ __launch_bounds__(256)
void LGNN_gemm(const __bf16* __restrict__ A, const __bf16* __restrict__ Wt,
               const float* __restrict__ bias, const float* __restrict__ addC,
               const int* __restrict__ gidx,
               float* __restrict__ outF, __bf16* __restrict__ outH,
               int M, int K, int Nout, int relu)
{
    int wave = blockIdx.x * (blockDim.x >> 5) + (threadIdx.x >> 5);
    int lane = threadIdx.x & 31;
    int tilesM = M >> 4;
    int tilesN = Nout >> 6;                         // 64 columns per wave
    if (wave >= tilesM * tilesN) return;            // wave-uniform guard
    int tm = wave % tilesM;
    int tn = wave / tilesM;

    int r16 = lane & 15;
    int hh  = lane >> 4;

    int m = tm * 16 + r16;
    long rowA = gidx ? (long)gidx[m] : (long)m;
    const __bf16* __restrict__ arow = A + rowA * (long)K;

    const __bf16* brow[4];
#pragma unroll
    for (int t = 0; t < 4; ++t)
        brow[t] = Wt + (long)(tn * 64 + t * 16 + r16) * K;

    v8f acc[4];
#pragma unroll
    for (int t = 0; t < 4; ++t)
        acc[t] = (v8f){0.f, 0.f, 0.f, 0.f, 0.f, 0.f, 0.f, 0.f};

    for (int kk = 0; kk < K; kk += 32) {
        // WGP-scope prefetch (locality 3): fill all cache levels on miss
        __builtin_prefetch(arow + kk + 128, 0, 3);          // global_prefetch_b8
        BF16Frag a;
        a.h[0] = *(const v8bf*)(arow + kk + 8 * hh);        // global_load_b128
        a.h[1] = *(const v8bf*)(arow + kk + 16 + 8 * hh);
#pragma unroll
        for (int t = 0; t < 4; ++t) {
            BF16Frag b;
            b.h[0] = *(const v8bf*)(brow[t] + kk + 8 * hh);
            b.h[1] = *(const v8bf*)(brow[t] + kk + 16 + 8 * hh);
            acc[t] = __builtin_amdgcn_wmma_f32_16x16x32_bf16(
                false, a.f, false, b.f, (short)0, acc[t], false, false);
        }
    }

#pragma unroll
    for (int t = 0; t < 4; ++t) {
        int n = tn * 64 + t * 16 + r16;
        float bn = bias ? bias[n] : 0.f;
#pragma unroll
        for (int r = 0; r < 8; ++r) {
            int mo = tm * 16 + r + 8 * hh;          // C/D: vgpr r -> row r+8*hh
            float v = acc[t][r] + bn;
            if (addC) v += addC[(long)mo * Nout + n];
            if (relu) v = fmaxf(v, 0.f);
            if (outH) outH[(long)mo * Nout + n] = (__bf16)v;
            else      outF[(long)mo * Nout + n] = v;
        }
    }
}

// f32 -> bf16, 4 elements per thread (float4 load -> packed 8B bf16 store)
__global__ void LGNN_cvt_bf16(const float* __restrict__ s, __bf16* __restrict__ d, long n4)
{
    long i = (long)blockIdx.x * blockDim.x + threadIdx.x;
    long st = (long)gridDim.x * blockDim.x;
    const float4* s4 = (const float4*)s;
    for (; i < n4; i += st) {
        float4 v = s4[i];
        v4bf o;
        o[0] = (__bf16)v.x; o[1] = (__bf16)v.y;
        o[2] = (__bf16)v.z; o[3] = (__bf16)v.w;
        *(v4bf*)(d + 4 * i) = o;
    }
}

// Weight pack: Wt[n*K + k] = (bf16) W[k*Nout + n]   (transpose + convert)
__global__ void LGNN_wt_pack(const float* __restrict__ W, __bf16* __restrict__ Wt,
                             int K, int Nout)
{
    long i = (long)blockIdx.x * blockDim.x + threadIdx.x;
    if (i >= (long)K * Nout) return;
    long n = i / K;
    long k = i - n * K;
    Wt[i] = (__bf16)W[k * (long)Nout + n];
}

// ---------------------------------------------------------------------------
// Node attention: one wave per (edge, head). 64 dims -> 2 per lane.
// ---------------------------------------------------------------------------
__global__ __launch_bounds__(256)
void LGNN_node_attn(const float* __restrict__ q, const float* __restrict__ k,
                    const float* __restrict__ v, const float* __restrict__ lg,
                    const int* __restrict__ src, const int* __restrict__ dst,
                    float* __restrict__ wv, float* __restrict__ z)
{
    int gw = blockIdx.x * (blockDim.x >> 5) + (threadIdx.x >> 5);
    int lane = threadIdx.x & 31;
    if (gw >= N_EDGES * NHEAD) return;
    int e  = gw >> 3;
    int hh = gw & 7;
    int s = src[e], d = dst[e];

    const float* kp = k  + (long)s * NDIM + hh * HDIM;
    const float* qp = q  + (long)d * NDIM + hh * HDIM;
    const float* vp = v  + (long)s * NDIM + hh * HDIM;
    const float* ep = lg + (long)e * HDIM;

    float k0 = kp[lane]      + ep[lane];
    float k1 = kp[lane + 32] + ep[lane + 32];
    float acc = k0 * qp[lane] + k1 * qp[lane + 32];
#pragma unroll
    for (int o = 16; o > 0; o >>= 1) acc += __shfl_xor(acc, o, 32);

    float sc = __expf(fminf(fmaxf(acc * 0.125f, -10.f), 10.f));

    float* wp = wv + (long)d * NDIM + hh * HDIM;
    atomicAdd(&wp[lane],      (vp[lane]      + ep[lane])      * sc);
    atomicAdd(&wp[lane + 32], (vp[lane + 32] + ep[lane + 32]) * sc);
    if (lane == 0) atomicAdd(&z[d * NHEAD + hh], sc);
}

// Edge (line-graph) attention: one wave per lg-edge, 64-dim single head.
__global__ __launch_bounds__(256)
void LGNN_edge_attn(const float* __restrict__ q, const float* __restrict__ k,
                    const float* __restrict__ v, const int* __restrict__ lsrc,
                    const int* __restrict__ ldst, float* __restrict__ wv,
                    float* __restrict__ z)
{
    int gw = blockIdx.x * (blockDim.x >> 5) + (threadIdx.x >> 5);
    int lane = threadIdx.x & 31;
    if (gw >= N_LGE) return;
    int s = lsrc[gw], d = ldst[gw];

    const float* kp = k + (long)s * HDIM;
    const float* qp = q + (long)d * HDIM;
    const float* vp = v + (long)s * HDIM;

    float acc = kp[lane] * qp[lane] + kp[lane + 32] * qp[lane + 32];
#pragma unroll
    for (int o = 16; o > 0; o >>= 1) acc += __shfl_xor(acc, o, 32);

    float sc = __expf(fminf(fmaxf(acc * 0.125f, -10.f), 10.f));

    float* wp = wv + (long)d * HDIM;
    atomicAdd(&wp[lane],      vp[lane] * sc);
    atomicAdd(&wp[lane + 32], vp[lane + 32] * sc);
    if (lane == 0) atomicAdd(&z[d], sc);
}

// o = wv / z
__global__ void LGNN_div_node(const float* __restrict__ wv, const float* __restrict__ z,
                              float* __restrict__ o)
{
    long i = (long)blockIdx.x * blockDim.x + threadIdx.x;
    if (i >= (long)N_NODES * NDIM) return;
    int n  = (int)(i >> 9);
    int hh = (int)((i >> 6) & 7);
    o[i] = wv[i] / z[n * NHEAD + hh];
}

__global__ void LGNN_div_edge(const float* __restrict__ wv, const float* __restrict__ z,
                              float* __restrict__ o)
{
    long i = (long)blockIdx.x * blockDim.x + threadIdx.x;
    if (i >= (long)N_EDGES * HDIM) return;
    o[i] = wv[i] / z[i >> 6];
}

// out = LayerNorm(a + b) * gamma + beta, one wave per row, D in {64, 512}
__global__ __launch_bounds__(256)
void LGNN_res_ln(const float* __restrict__ a, const float* __restrict__ b,
                 const float* __restrict__ g, const float* __restrict__ bt,
                 float* __restrict__ out, int rows, int D)
{
    int row = blockIdx.x * (blockDim.x >> 5) + (threadIdx.x >> 5);
    int lane = threadIdx.x & 31;
    if (row >= rows) return;
    int cnt = D >> 5;
    const float* ap = a + (long)row * D;
    const float* bp = b + (long)row * D;

    float t[16];
    float s = 0.f, s2 = 0.f;
    for (int i = 0; i < cnt; ++i) {
        float x = ap[lane + 32 * i] + bp[lane + 32 * i];
        t[i] = x; s += x; s2 += x * x;
    }
#pragma unroll
    for (int o = 16; o > 0; o >>= 1) { s += __shfl_xor(s, o, 32); s2 += __shfl_xor(s2, o, 32); }
    float mean = s / (float)D;
    float var  = s2 / (float)D - mean * mean;
    float rs   = rsqrtf(var + 1e-5f);
    for (int i = 0; i < cnt; ++i) {
        int d = lane + 32 * i;
        out[(long)row * D + d] = (t[i] - mean) * rs * g[d] + bt[d];
    }
}

__global__ void LGNN_rel_gather(const float* __restrict__ rel, const int* __restrict__ feat,
                                float* __restrict__ lg)
{
    long i = (long)blockIdx.x * blockDim.x + threadIdx.x;
    if (i >= (long)N_EDGES * HDIM) return;
    lg[i] = rel[(long)feat[i >> 6] * HDIM + (i & 63)];
}

__global__ void LGNN_zero(float* __restrict__ p, long n)
{
    long i = (long)blockIdx.x * blockDim.x + threadIdx.x;
    long st = (long)gridDim.x * blockDim.x;
    for (; i < n; i += st) p[i] = 0.f;
}

__global__ void LGNN_copy(const float* __restrict__ s, float* __restrict__ d, long n)
{
    long i = (long)blockIdx.x * blockDim.x + threadIdx.x;
    long st = (long)gridDim.x * blockDim.x;
    for (; i < n; i += st) d[i] = s[i];
}

// ---------------------------------------------------------------------------
// Host-side orchestration
// ---------------------------------------------------------------------------
static inline void gemm(const __bf16* A, const __bf16* Wt, const float* bias,
                        const float* addC, const int* gidx, float* outF, __bf16* outH,
                        int M, int K, int Nout, int relu, hipStream_t st)
{
    int tiles = (M >> 4) * (Nout >> 6);
    LGNN_gemm<<<(tiles + 7) / 8, 256, 0, st>>>(A, Wt, bias, addC, gidx, outF, outH,
                                               M, K, Nout, relu);
}

static inline void cvt(const float* s, __bf16* d, long n, hipStream_t st)
{
    LGNN_cvt_bf16<<<2048, 256, 0, st>>>(s, d, n >> 2);   // n divisible by 4
}

static inline void zero(float* p, long n, hipStream_t st)
{
    LGNN_zero<<<2048, 256, 0, st>>>(p, n);
}

extern "C" void kernel_launch(void* const* d_in, const int* in_sizes, int n_in,
                              void* d_out, int out_size, void* d_ws, size_t ws_size,
                              hipStream_t stream)
{
    (void)in_sizes; (void)out_size; (void)ws_size;
    if (n_in < 67) return;

    const float* x_in      = (const float*)d_in[0];
    const int*   edge_feat = (const int*)d_in[1];
    const int*   src_ids   = (const int*)d_in[2];
    const int*   dst_ids   = (const int*)d_in[3];
    const int*   lg_src    = (const int*)d_in[4];
    const int*   lg_dst    = (const int*)d_in[5];

    // JAX pytree flatten order (dict keys sorted):
    //  edge keys: e,f1,f2,k,ln1,ln2,q,v -> 16 leaves; node: f1,f2,k,ln1,ln2,q,v -> 14
    auto F = [&](int i) { return (const float*)d_in[i]; };
    const int eb[2] = {6, 36};
    const int nb[2] = {22, 52};
    const float* rel = F(66);

    // ---- workspace layout ----
    float* ws = (float*)d_ws;
    size_t off = 0;
    float* xA  = ws + off; off += (size_t)N_NODES * NDIM;
    float* xB  = ws + off; off += (size_t)N_NODES * NDIM;
    float* lgA = ws + off; off += (size_t)N_EDGES * HDIM;
    float* lgB = ws + off; off += (size_t)N_EDGES * HDIM;
    float* scr = ws + off;
    // f32 union scratch (node phase then edge phase)
    float* qb  = scr;
    float* kb  = qb  + (size_t)N_NODES * NDIM;
    float* vb  = kb  + (size_t)N_NODES * NDIM;
    float* wvb = vb  + (size_t)N_NODES * NDIM;
    float* zb  = wvb + (size_t)N_NODES * NDIM;
    float* hb  = zb  + (size_t)N_NODES * NHEAD;
    float* f2b = hb  + (size_t)N_NODES * NDIM;

    float* ee   = scr;
    float* qe   = ee   + (size_t)N_EDGES * HDIM;
    float* ke   = qe   + (size_t)N_EDGES * HDIM;
    float* ve   = ke   + (size_t)N_EDGES * HDIM;
    float* wve  = ve   + (size_t)N_EDGES * HDIM;
    float* ze   = wve  + (size_t)N_EDGES * HDIM;
    float* he   = ze   + (size_t)N_EDGES;
    float* f2e  = he   + (size_t)N_EDGES * HDIM;
    size_t scr_f32 = (size_t)N_EDGES * (HDIM * 7 + 1);   // edge phase is the max
    off += scr_f32;

    // bf16 arena (16B aligned: all offsets are multiples of 8 floats)
    __bf16* harena = (__bf16*)(ws + off);
    __bf16* hp = harena;
    // packed transposed weights, per layer
    __bf16 *WqN[2], *WkN[2], *WvN[2], *W1N[2], *W2N[2];
    __bf16 *WeE[2], *WqE[2], *WkE[2], *WvE[2], *W1E[2], *W2E[2];
    for (int l = 0; l < 2; ++l) {
        WqN[l] = hp; hp += (size_t)NDIM * NDIM;
        WkN[l] = hp; hp += (size_t)NDIM * NDIM;
        WvN[l] = hp; hp += (size_t)NDIM * NDIM;
        W1N[l] = hp; hp += (size_t)NDIM * FF_N;
        W2N[l] = hp; hp += (size_t)FF_N * NDIM;
        WeE[l] = hp; hp += (size_t)NDIM * HDIM;
        WqE[l] = hp; hp += (size_t)HDIM * HDIM;
        WkE[l] = hp; hp += (size_t)HDIM * HDIM;
        WvE[l] = hp; hp += (size_t)HDIM * HDIM;
        W1E[l] = hp; hp += (size_t)HDIM * FF_E;
        W2E[l] = hp; hp += (size_t)FF_E * HDIM;
    }
    __bf16* abuf   = hp; hp += (size_t)N_EDGES * HDIM;   // staged A (max 10.24M elems)
    __bf16* midb16 = hp;                                  // FFN mid, max E*256 elems

    // ---- pack weights (transpose + bf16) ----
    auto pack = [&](const float* W, __bf16* Wt, int K, int Nt) {
        long n = (long)K * Nt;
        LGNN_wt_pack<<<(int)((n + 255) / 256), 256, 0, stream>>>(W, Wt, K, Nt);
    };
    for (int l = 0; l < 2; ++l) {
        const int e0 = eb[l], n0 = nb[l];
        pack(F(n0 + 10), WqN[l], NDIM, NDIM);
        pack(F(n0 + 4),  WkN[l], NDIM, NDIM);
        pack(F(n0 + 12), WvN[l], NDIM, NDIM);
        pack(F(n0 + 0),  W1N[l], NDIM, FF_N);
        pack(F(n0 + 2),  W2N[l], FF_N, NDIM);
        pack(F(e0 + 0),  WeE[l], NDIM, HDIM);
        pack(F(e0 + 12), WqE[l], HDIM, HDIM);
        pack(F(e0 + 6),  WkE[l], HDIM, HDIM);
        pack(F(e0 + 14), WvE[l], HDIM, HDIM);
        pack(F(e0 + 2),  W1E[l], HDIM, FF_E);
        pack(F(e0 + 4),  W2E[l], FF_E, HDIM);
    }

    // ---- init running state ----
    LGNN_copy<<<2048, 256, 0, stream>>>(x_in, xA, (long)N_NODES * NDIM);
    {
        long n = (long)N_EDGES * HDIM;
        LGNN_rel_gather<<<(int)((n + 255) / 256), 256, 0, stream>>>(rel, edge_feat, lgA);
    }

    float* xcur = xA;  float* xnxt = xB;
    float* lgcur = lgA; float* lgnxt = lgB;

    for (int l = 0; l < 2; ++l) {
        const int e0 = eb[l], n0 = nb[l];
        const float *nb1 = F(n0 + 1), *nb2 = F(n0 + 3), *nbk = F(n0 + 5);
        const float *ng1 = F(n0 + 6), *nB1 = F(n0 + 7), *ng2 = F(n0 + 8), *nB2 = F(n0 + 9);
        const float *nbq = F(n0 + 11), *nbv = F(n0 + 13);
        const float *ebe = F(e0 + 1), *eb1 = F(e0 + 3), *eb2 = F(e0 + 5), *ebk = F(e0 + 7);
        const float *eg1 = F(e0 + 8), *eB1 = F(e0 + 9), *eg2 = F(e0 + 10), *eB2 = F(e0 + 11);
        const float *ebq = F(e0 + 13), *ebv = F(e0 + 15);

        // ===== node update (old x, old lg) -> xnxt =====
        cvt(xcur, abuf, (long)N_NODES * NDIM, stream);
        gemm(abuf, WqN[l], nbq, nullptr, nullptr, qb, nullptr, N_NODES, NDIM, NDIM, 0, stream);
        gemm(abuf, WkN[l], nbk, nullptr, nullptr, kb, nullptr, N_NODES, NDIM, NDIM, 0, stream);
        gemm(abuf, WvN[l], nbv, nullptr, nullptr, vb, nullptr, N_NODES, NDIM, NDIM, 0, stream);
        zero(wvb, (long)N_NODES * NDIM, stream);
        zero(zb,  (long)N_NODES * NHEAD, stream);
        {
            int waves = N_EDGES * NHEAD;
            LGNN_node_attn<<<(waves + 7) / 8, 256, 0, stream>>>(qb, kb, vb, lgcur,
                                                                src_ids, dst_ids, wvb, zb);
        }
        {
            long n = (long)N_NODES * NDIM;      // o reuses qb
            LGNN_div_node<<<(int)((n + 255) / 256), 256, 0, stream>>>(wvb, zb, qb);
        }
        LGNN_res_ln<<<(N_NODES + 7) / 8, 256, 0, stream>>>(xcur, qb, ng1, nB1, hb, N_NODES, NDIM);
        cvt(hb, abuf, (long)N_NODES * NDIM, stream);
        gemm(abuf, W1N[l], nb1, nullptr, nullptr, nullptr, midb16, N_NODES, NDIM, FF_N, 1, stream);
        gemm(midb16, W2N[l], nb2, nullptr, nullptr, f2b, nullptr, N_NODES, FF_N, NDIM, 0, stream);
        LGNN_res_ln<<<(N_NODES + 7) / 8, 256, 0, stream>>>(hb, f2b, ng2, nB2, xnxt, N_NODES, NDIM);

        // ===== edge update (old lg, old x gathered by dst) -> lgnxt =====
        cvt(xcur, abuf, (long)N_NODES * NDIM, stream);
        gemm(abuf, WeE[l], ebe, nullptr, dst_ids, ee, nullptr, N_EDGES, NDIM, HDIM, 0, stream);
        cvt(lgcur, abuf, (long)N_EDGES * HDIM, stream);
        gemm(abuf, WqE[l], ebq, nullptr, nullptr, qe, nullptr, N_EDGES, HDIM, HDIM, 0, stream);
        gemm(abuf, WkE[l], ebk, ee, nullptr, ke, nullptr, N_EDGES, HDIM, HDIM, 0, stream);
        gemm(abuf, WvE[l], ebv, ee, nullptr, ve, nullptr, N_EDGES, HDIM, HDIM, 0, stream);
        zero(wve, (long)N_EDGES * HDIM, stream);
        zero(ze,  (long)N_EDGES, stream);
        LGNN_edge_attn<<<(N_LGE + 7) / 8, 256, 0, stream>>>(qe, ke, ve, lg_src, lg_dst, wve, ze);
        {
            long n = (long)N_EDGES * HDIM;      // o reuses qe
            LGNN_div_edge<<<(int)((n + 255) / 256), 256, 0, stream>>>(wve, ze, qe);
        }
        LGNN_res_ln<<<(N_EDGES + 7) / 8, 256, 0, stream>>>(lgcur, qe, eg1, eB1, he, N_EDGES, HDIM);
        cvt(he, abuf, (long)N_EDGES * HDIM, stream);
        gemm(abuf, W1E[l], eb1, nullptr, nullptr, nullptr, midb16, N_EDGES, HDIM, FF_E, 1, stream);
        gemm(midb16, W2E[l], eb2, nullptr, nullptr, f2e, nullptr, N_EDGES, FF_E, HDIM, 0, stream);
        LGNN_res_ln<<<(N_EDGES + 7) / 8, 256, 0, stream>>>(he, f2e, eg2, eB2, lgnxt, N_EDGES, HDIM);

        // swap ping-pong
        float* t;
        t = xcur; xcur = xnxt; xnxt = t;
        t = lgcur; lgcur = lgnxt; lgnxt = t;
    }

    // ---- outputs: (x, lg_x) concatenated flat ----
    float* out = (float*)d_out;
    LGNN_copy<<<2048, 256, 0, stream>>>(xcur, out, (long)N_NODES * NDIM);
    LGNN_copy<<<2048, 256, 0, stream>>>(lgcur, out + (size_t)N_NODES * NDIM,
                                        (long)N_EDGES * HDIM);
}
